// LSTM_80702435492285
// MI455X (gfx1250) — matrix operand order repaired and checked by reading, not compiled, not run on hardware
//
#include <hip/hip_runtime.h>
#include <hip/hip_bf16.h>

// ---------------------------------------------------------------------------
// LSTM on MI455X (gfx1250): bf16 WMMA recurrence, weights staged in 256KB LDS,
// register-resident cell state, persistent cooperative kernel.
// ---------------------------------------------------------------------------

typedef __attribute__((ext_vector_type(16))) __bf16 v16bf;
typedef __attribute__((ext_vector_type(8)))  __bf16 v8bf;
typedef __attribute__((ext_vector_type(8)))  float  v8f;

#define NB      64       // batch
#define T_STEPS 1024
#define DIM_D   512
#define DIM_H   512
#define DIM_F   2048     // 4*H
#define KTOT    1024     // D + H (fused [x_t ; h] GEMM)

#define NWG     16       // persistent workgroups (1 per 2 hidden tiles)
#define WAVES   8        // waves per workgroup: 4 batch-tiles x 2 hidden-tiles

#define LDS_BYTES (2 * 4 * 32 * 512 * 2)   // 2 htl * 4 gates * 32 ksteps * 1KB frag = 256KB

// workspace layout (bytes)
#define XB_BYTES   ((size_t)NB * T_STEPS * DIM_D * 2)        // 64 MiB  bf16 x
#define WC_BYTES   ((size_t)DIM_F * KTOT * 2)                // 4 MiB   bf16 [Wx;Wh]^T
#define HB_BYTES   ((size_t)2 * NB * DIM_H * 2)              // 128 KiB ping-pong h (bf16)

__device__ __forceinline__ float fast_sigmoid(float x) { return 1.0f / (1.0f + __expf(-x)); }
__device__ __forceinline__ float fast_tanh(float x)    { return 2.0f / (1.0f + __expf(-2.0f * x)) - 1.0f; }

// A fragment (16x32 bf16, ISA 16-bit A layout):
// lane<16 : K = [0..8) and [16..24) of the tile ; lane>=16 : +8 (folded into p)
__device__ __forceinline__ v16bf load_fragA(const __bf16* p) {
    v8bf lo = *(const v8bf*)(p);
    v8bf hi = *(const v8bf*)(p + 16);
    v16bf r;
#pragma unroll
    for (int i = 0; i < 8; ++i) { r[i] = lo[i]; r[i + 8] = hi[i]; }
    return r;
}

// Agent-scope generation barrier across NWG blocks.
__device__ __forceinline__ void grid_sync(unsigned* bar) {
    __syncthreads();
    __threadfence();                       // release h stores to device scope
    if (threadIdx.x == 0) {
        unsigned* cnt = bar;
        unsigned* gen = bar + 32;          // separate cacheline
        unsigned g = __hip_atomic_load(gen, __ATOMIC_RELAXED, __HIP_MEMORY_SCOPE_AGENT);
        unsigned old = __hip_atomic_fetch_add(cnt, 1u, __ATOMIC_ACQ_REL, __HIP_MEMORY_SCOPE_AGENT);
        if (old == NWG - 1u) {
            __hip_atomic_store(cnt, 0u, __ATOMIC_RELAXED, __HIP_MEMORY_SCOPE_AGENT);
            __hip_atomic_fetch_add(gen, 1u, __ATOMIC_ACQ_REL, __HIP_MEMORY_SCOPE_AGENT);
        } else {
            while (__hip_atomic_load(gen, __ATOMIC_ACQUIRE, __HIP_MEMORY_SCOPE_AGENT) == g) {
                __builtin_amdgcn_s_sleep(1);
            }
        }
    }
    __syncthreads();
    __threadfence();                       // acquire side for all threads
}

// -------------------------- prep kernels -----------------------------------

// x (f32) -> xb (bf16), identical [n][t][d] layout (x is a read-once stream)
__global__ void k_xcvt(const float* __restrict__ x, __bf16* __restrict__ xb) {
    size_t i = (size_t)blockIdx.x * blockDim.x + threadIdx.x;
    if (i < (size_t)NB * T_STEPS * DIM_D)
        xb[i] = (__bf16)__builtin_nontemporal_load(x + i);
}

// Wc[f][k] = (k<512 ? Wx[k][f] : Wh[k-512][f]) as bf16  (B^T, row pitch KTOT)
__global__ void k_wcvt(const float* __restrict__ Wx, const float* __restrict__ Wh,
                       __bf16* __restrict__ Wc) {
    size_t idx = (size_t)blockIdx.x * blockDim.x + threadIdx.x;
    if (idx >= (size_t)DIM_F * KTOT) return;
    int f = (int)(idx >> 10);      // 0..2047
    int k = (int)(idx & 1023);     // 0..1023
    float v = (k < DIM_D) ? Wx[(size_t)k * DIM_F + f]
                          : Wh[(size_t)(k - DIM_D) * DIM_F + f];
    Wc[idx] = (__bf16)v;
}

// h0 -> hb buffer 0 (bf16); zero barrier state
__global__ void k_init(const float* __restrict__ h0, __bf16* __restrict__ hb,
                       unsigned* __restrict__ bar) {
    int i = blockIdx.x * blockDim.x + threadIdx.x;
    if (i < NB * DIM_H) hb[i] = (__bf16)h0[i];
    if (i < 64) bar[i] = 0u;
}

// -------------------------- recurrent kernel -------------------------------

__global__ void __launch_bounds__(256, 1)
lstm_recurrent(const __bf16* __restrict__ xb, const __bf16* __restrict__ Wc,
               __bf16* __restrict__ hb, const float* __restrict__ bias,
               float* __restrict__ out, unsigned* __restrict__ bar) {
    extern __shared__ __align__(128) __bf16 smem[];   // 256 KiB weight fragments

    const int tid  = threadIdx.x;
    const int lane = tid & 31;
    const int wv   = tid >> 5;                  // 0..7
    const int mt   = wv & 3;                    // batch tile 0..3
    const int htl  = wv >> 2;                   // local hidden tile 0..1
    const int ht   = blockIdx.x * 2 + htl;      // global hidden tile 0..31
    const int ln   = lane & 15;
    const int hi   = (lane >> 4) & 1;

    // ---- one-time: stage this block's weight fragments into LDS -----------
    // fragment fidx = fhtl*128 + g*32 + ks ; stored lane-major: 32B per lane.
    for (int e = tid; e < 256 * 32; e += 256) {
        const int fidx = e >> 5;
        const int l    = e & 31;
        const int fhtl = fidx >> 7;
        const int g    = (fidx >> 5) & 3;
        const int ks   = fidx & 31;
        const int row  = g * DIM_H + (blockIdx.x * 2 + fhtl) * 16 + (l & 15);
        const int col  = ks * 32 + ((l & 16) ? 8 : 0);
        v8bf wlo = *(const v8bf*)(Wc + (size_t)row * KTOT + col);
        v8bf whi = *(const v8bf*)(Wc + (size_t)row * KTOT + col + 16);
        v16bf wv16;
#pragma unroll
        for (int i = 0; i < 8; ++i) { wv16[i] = wlo[i]; wv16[i + 8] = whi[i]; }
        *(v16bf*)(smem + (size_t)fidx * 512 + l * 16) = wv16;
    }
    __syncthreads();

    const __bf16* smB  = smem + (size_t)htl * 128 * 512;   // this wave's 128 frags
    const __bf16* smBl = smB + lane * 16;                  // per-lane base

    // per-lane bias (depends only on output column)
    float bval[4];
#pragma unroll
    for (int g = 0; g < 4; ++g) bval[g] = bias[g * DIM_H + ht * 16 + ln];

    const int m = mt * 16 + ln;                             // batch row (A layout)
    const __bf16* xrowb = xb + (size_t)m * T_STEPS * DIM_D; // row base (no hi offset)
    const __bf16* xrow0 = xrowb + hi * 8;
    const __bf16* hrow0 = hb + (size_t)m * DIM_H + hi * 8;  // + parity*NB*H

    v8f c = {};   // cell state: stays in VGPRs for all 1024 steps

    const int nbb = mt * 16 + hi * 8;          // C/D row base for this half-wave
    const int col = ht * 16 + ln;              // C/D column

    for (int t = 0; t < T_STEPS; ++t) {
        v8f a0, a1, a2, a3;
#pragma unroll
        for (int r = 0; r < 8; ++r) { a0[r] = bval[0]; a1[r] = bval[1]; a2[r] = bval[2]; a3[r] = bval[3]; }

        // ---- x part: K = 0..511 (frag ks = 0..15 within each gate) ----
        const __bf16* xrow = xrow0 + (size_t)t * DIM_D;
#pragma unroll 4
        for (int ks = 0; ks < 16; ++ks) {
            v16bf av = load_fragA(xrow + ks * 32);
            v16bf b0 = *(const v16bf*)(smBl + (size_t)(0 * 32 + ks) * 512);
            v16bf b1 = *(const v16bf*)(smBl + (size_t)(1 * 32 + ks) * 512);
            v16bf b2 = *(const v16bf*)(smBl + (size_t)(2 * 32 + ks) * 512);
            v16bf b3 = *(const v16bf*)(smBl + (size_t)(3 * 32 + ks) * 512);
            a0 = __builtin_amdgcn_wmma_f32_16x16x32_bf16(false, av, false, b0, (short)0, a0, false, false);
            a1 = __builtin_amdgcn_wmma_f32_16x16x32_bf16(false, av, false, b1, (short)0, a1, false, false);
            a2 = __builtin_amdgcn_wmma_f32_16x16x32_bf16(false, av, false, b2, (short)0, a2, false, false);
            a3 = __builtin_amdgcn_wmma_f32_16x16x32_bf16(false, av, false, b3, (short)0, a3, false, false);
        }

        // ---- prefetch next step's x slab (arrives during h phase / barrier) ----
        {
            const __bf16* xnext = xrowb + (size_t)(t + 1) * DIM_D;
#pragma unroll
            for (int j = 0; j < 4; ++j)
                __builtin_prefetch(xnext + (hi * 4 + j) * 64, 0, 3);   // 4 x 128B lines/lane
        }

        // ---- h part: K = 512..1023 (frag ks = 16..31 within each gate) ----
        const __bf16* hrow = hrow0 + (size_t)(t & 1) * NB * DIM_H;
#pragma unroll 4
        for (int ks = 0; ks < 16; ++ks) {
            v16bf av = load_fragA(hrow + ks * 32);
            v16bf b0 = *(const v16bf*)(smBl + (size_t)(0 * 32 + 16 + ks) * 512);
            v16bf b1 = *(const v16bf*)(smBl + (size_t)(1 * 32 + 16 + ks) * 512);
            v16bf b2 = *(const v16bf*)(smBl + (size_t)(2 * 32 + 16 + ks) * 512);
            v16bf b3 = *(const v16bf*)(smBl + (size_t)(3 * 32 + 16 + ks) * 512);
            a0 = __builtin_amdgcn_wmma_f32_16x16x32_bf16(false, av, false, b0, (short)0, a0, false, false);
            a1 = __builtin_amdgcn_wmma_f32_16x16x32_bf16(false, av, false, b1, (short)0, a1, false, false);
            a2 = __builtin_amdgcn_wmma_f32_16x16x32_bf16(false, av, false, b2, (short)0, a2, false, false);
            a3 = __builtin_amdgcn_wmma_f32_16x16x32_bf16(false, av, false, b3, (short)0, a3, false, false);
        }

        // ---- gates in-register; c stays resident; publish h ----
        __bf16* hwr = hb + (size_t)((t + 1) & 1) * NB * DIM_H;
#pragma unroll
        for (int r = 0; r < 8; ++r) {
            float iv = fast_sigmoid(a0[r]);
            float fv = fast_sigmoid(a1[r]);
            float ov = fast_sigmoid(a2[r]);
            float gv = fast_tanh(a3[r]);
            float cn = fv * c[r] + iv * gv;
            c[r] = cn;
            float hN = ov * fast_tanh(cn);
            const int nb = nbb + r;
            // out is a write-once 128MB stream: keep it out of L2 (TH=NT)
            __builtin_nontemporal_store(hN, &out[((size_t)nb * T_STEPS + t) * DIM_H + col]);
            hwr[(size_t)nb * DIM_H + col] = (__bf16)hN;
        }

        grid_sync(bar);
    }
}

// -------------------------- host launcher ----------------------------------

extern "C" void kernel_launch(void* const* d_in, const int* in_sizes, int n_in,
                              void* d_out, int out_size, void* d_ws, size_t ws_size,
                              hipStream_t stream) {
    const float* x  = (const float*)d_in[0];   // (64,1024,512)
    const float* h0 = (const float*)d_in[1];   // (64,512)
    const float* Wx = (const float*)d_in[2];   // (512,2048)
    const float* Wh = (const float*)d_in[3];   // (512,2048)
    const float* b  = (const float*)d_in[4];   // (2048,)
    float* out = (float*)d_out;                // (64,1024,512)

    char* ws = (char*)d_ws;
    __bf16*   xb  = (__bf16*)(ws);
    __bf16*   Wc  = (__bf16*)(ws + XB_BYTES);
    __bf16*   hb  = (__bf16*)(ws + XB_BYTES + WC_BYTES);
    unsigned* bar = (unsigned*)(ws + XB_BYTES + WC_BYTES + HB_BYTES);

    {   // x -> bf16
        size_t n = (size_t)NB * T_STEPS * DIM_D;
        k_xcvt<<<(unsigned)((n + 255) / 256), 256, 0, stream>>>(x, xb);
    }
    {   // [Wx;Wh]^T -> bf16
        size_t n = (size_t)DIM_F * KTOT;
        k_wcvt<<<(unsigned)((n + 255) / 256), 256, 0, stream>>>(Wx, Wh, Wc);
    }
    k_init<<<(NB * DIM_H + 255) / 256, 256, 0, stream>>>(h0, hb, bar);

    lstm_recurrent<<<NWG, WAVES * 32, LDS_BYTES, stream>>>(xb, Wc, hb, b, out, bar);
}
